// Edge_Cycle_50869592655535
// MI455X (gfx1250) — matrix operand order repaired and checked by reading, not compiled, not run on hardware
//
#include <hip/hip_runtime.h>

typedef __attribute__((ext_vector_type(16))) _Float16 v16h;
typedef __attribute__((ext_vector_type(8)))  _Float16 v8h;
typedef __attribute__((ext_vector_type(8)))  float    v8f;

#define BN_ROWS 512

// ---------------- utility kernels ----------------

__global__ void zero_f32_k(float* __restrict__ p, long long n) {
  long long i  = (long long)blockIdx.x * blockDim.x + threadIdx.x;
  long long st = (long long)gridDim.x * blockDim.x;
  for (; i < n; i += st) p[i] = 0.f;
}

__global__ void cvt_f16_k(const float* __restrict__ s, _Float16* __restrict__ d, int n) {
  int i = blockIdx.x * blockDim.x + threadIdx.x;
  if (i < n) d[i] = (_Float16)s[i];
}

// a[idx1[e]] += src[idx0[e]]   (128 channels, f32 atomics)
__global__ void scatter_add128_k(const float* __restrict__ src, const int* __restrict__ idx,
                                 int m, float* __restrict__ dst) {
  int e = blockIdx.x;
  int c = threadIdx.x;
  int s = idx[e];
  int t = idx[m + e];
  atomicAdd(dst + (size_t)t * 128 + c, src[(size_t)s * 128 + c]);
}

// per-group (k consecutive rows) column sums
__global__ void group_sum_k(const float* __restrict__ a, float* __restrict__ gs, int k) {
  int g = blockIdx.x, c = threadIdx.x;  // 128 threads
  const float* p = a + (size_t)g * k * 128 + c;
  float s = 0.f;
  for (int j = 0; j < k; ++j) s += p[(size_t)j * 128];
  gs[(size_t)g * 128 + c] = s;
}

// g16[j] = f16([a[j] | gs[j/k]])  (256 wide)
__global__ void pack_g16_k(const float* __restrict__ a, const float* __restrict__ gs,
                           _Float16* __restrict__ g16, int k) {
  int j = blockIdx.x, c = threadIdx.x;  // 256 threads
  float v = (c < 128) ? a[(size_t)j * 128 + c] : gs[(size_t)(j / k) * 128 + (c - 128)];
  g16[(size_t)j * 256 + c] = (_Float16)v;
}

// new16[j] = f16([gw[j] | rep[j]])
__global__ void pack_new16_k(const float* __restrict__ gw, const float* __restrict__ rep,
                             _Float16* __restrict__ n16) {
  int j = blockIdx.x, c = threadIdx.x;  // 256 threads
  float v = (c < 128) ? gw[(size_t)j * 128 + c] : rep[(size_t)j * 128 + (c - 128)];
  n16[(size_t)j * 256 + c] = (_Float16)v;
}

// dst[idx1[e]] += [gw|rep][idx0[e]]   (256 channels, f32 atomics)
__global__ void scatter_add_new_k(const float* __restrict__ gw, const float* __restrict__ rep,
                                  const int* __restrict__ idx, int m, float* __restrict__ dst) {
  int e = blockIdx.x, c = threadIdx.x;  // 256 threads
  int s = idx[e];
  int t = idx[m + e];
  float v = (c < 128) ? gw[(size_t)s * 128 + c] : rep[(size_t)s * 128 + (c - 128)];
  atomicAdd(dst + (size_t)t * 256 + c, v);
}

// X_e = f16([edge(128) | A_c(256) | pairsum(A_c)(256) | A_p(256) | pairsum(A_p)(256)])
__global__ void build_xe_k(const float* __restrict__ edge, const float* __restrict__ Ac,
                           const float* __restrict__ Ap, _Float16* __restrict__ X) {
  int j = blockIdx.x, c = threadIdx.x;  // 128 threads
  int jp = j ^ 1;                       // pair partner (k==2, consecutive groups)
  size_t base = (size_t)j * 1152;
  const float* acj = Ac + (size_t)j * 256;
  const float* acp = Ac + (size_t)jp * 256;
  const float* apj = Ap + (size_t)j * 256;
  const float* app = Ap + (size_t)jp * 256;
  X[base + c]        = (_Float16)edge[(size_t)j * 128 + c];
  X[base + 128 + c]  = (_Float16)acj[c];
  X[base + 256 + c]  = (_Float16)acj[128 + c];
  X[base + 384 + c]  = (_Float16)(acj[c] + acp[c]);
  X[base + 512 + c]  = (_Float16)(acj[128 + c] + acp[128 + c]);
  X[base + 640 + c]  = (_Float16)apj[c];
  X[base + 768 + c]  = (_Float16)apj[128 + c];
  X[base + 896 + c]  = (_Float16)(apj[c] + app[c]);
  X[base + 1024 + c] = (_Float16)(apj[128 + c] + app[128 + c]);
}

// ---------------- WMMA f16->f32 GEMM: C[M,N] = A16[M,K] x B16[K,N] ----------------
// One wave computes a 16x64 C tile (4 wmma accumulators), K stepped by 32.
// Fragment layouts per CDNA5 ISA 7.12.2 (wave32):
//   A 16x32: lane&15 = M row; lanes>=16 take K+8; halves 0..7 -> K{0..7}, 8..15 -> K{16..23}
//   B 32x16: lane&15 = N col; lanes>=16 take K+8; halves 0..7 -> K{0..7}, 8..15 -> K{16..23} (strided gather)
//   C/D:     VGPR i -> row i + 8*(lane>=16), col = lane&15
__global__ __launch_bounds__(256) void wmma_gemm_k(const _Float16* __restrict__ A,
                                                   const _Float16* __restrict__ B,
                                                   float* __restrict__ C,
                                                   int M, int N, int K, int numNT) {
  int gw   = (int)((blockIdx.x * blockDim.x + threadIdx.x) >> 5);
  int lane = threadIdx.x & 31;
  int numMT = (M + 15) >> 4;
  if (gw >= numMT * numNT) return;
  int mt = gw / numNT;
  int nt = gw - mt * numNT;
  int m0 = mt << 4;
  int n0 = nt << 6;

  int rowA = m0 + (lane & 15);
  if (rowA >= M) rowA = M - 1;              // tail: clamp (results masked on store)
  const _Float16* Ar = A + (size_t)rowA * K;
  int hiSel = (lane & 16) >> 1;             // 0 or 8

  v8f acc0 = {}, acc1 = {}, acc2 = {}, acc3 = {};
  for (int k0 = 0; k0 < K; k0 += 32) {
    union { v16h v; v8h h[2]; } af;
    af.h[0] = *(const v8h*)(Ar + k0 + hiSel);
    af.h[1] = *(const v8h*)(Ar + k0 + 16 + hiSel);
    const _Float16* Br = B + (size_t)(k0 + hiSel) * N + n0 + (lane & 15);
    v16h b0, b1, b2, b3;
#pragma unroll
    for (int s = 0; s < 16; ++s) {
      const _Float16* Bk = Br + (size_t)(s < 8 ? s : s + 8) * N;
      b0[s] = Bk[0];
      b1[s] = Bk[16];
      b2[s] = Bk[32];
      b3[s] = Bk[48];
    }
    acc0 = __builtin_amdgcn_wmma_f32_16x16x32_f16(false, af.v, false, b0, (short)0, acc0, false, false);
    acc1 = __builtin_amdgcn_wmma_f32_16x16x32_f16(false, af.v, false, b1, (short)0, acc1, false, false);
    acc2 = __builtin_amdgcn_wmma_f32_16x16x32_f16(false, af.v, false, b2, (short)0, acc2, false, false);
    acc3 = __builtin_amdgcn_wmma_f32_16x16x32_f16(false, af.v, false, b3, (short)0, acc3, false, false);
  }

  int rb = m0 + ((lane & 16) >> 1);
  int cb = n0 + (lane & 15);
#pragma unroll
  for (int i = 0; i < 8; ++i) {
    int rr = rb + i;
    if (rr < M) {
      float* cr = C + (size_t)rr * N + cb;
      cr[0]  = acc0[i];
      cr[16] = acc1[i];
      cr[32] = acc2[i];
      cr[48] = acc3[i];
    }
  }
}

// ---------------- BatchNorm (per-matrix, axis 0) ----------------

__global__ void bn_stats_k(const float* __restrict__ y, float* __restrict__ stats, int M, int C) {
  int c  = threadIdx.x;  // C threads
  int r0 = blockIdx.x * BN_ROWS;
  int r1 = r0 + BN_ROWS; if (r1 > M) r1 = M;
  float s = 0.f, s2 = 0.f;
  for (int rr = r0; rr < r1; ++rr) {
    float v = y[(size_t)rr * C + c];
    s += v; s2 += v * v;
  }
  atomicAdd(stats + c, s);
  atomicAdd(stats + C + c, s2);
}

__global__ void bn_relu_f16_k(const float* __restrict__ y, const float* __restrict__ stats,
                              const float* __restrict__ g, const float* __restrict__ b,
                              _Float16* __restrict__ o, int M, int C) {
  int c = threadIdx.x;
  float inv  = 1.f / (float)M;
  float mean = stats[c] * inv;
  float var  = stats[C + c] * inv - mean * mean;
  float sc   = g[c] * rsqrtf(var + 1e-5f);
  float sh   = b[c] - mean * sc;
  int r0 = blockIdx.x * BN_ROWS;
  int r1 = r0 + BN_ROWS; if (r1 > M) r1 = M;
  for (int rr = r0; rr < r1; ++rr) {
    float v = fmaf(y[(size_t)rr * C + c], sc, sh);
    o[(size_t)rr * C + c] = (_Float16)fmaxf(v, 0.f);
  }
}

__global__ void bn_relu_f32_k(const float* __restrict__ y, const float* __restrict__ stats,
                              const float* __restrict__ g, const float* __restrict__ b,
                              float* __restrict__ o, int M, int C) {
  int c = threadIdx.x;
  float inv  = 1.f / (float)M;
  float mean = stats[c] * inv;
  float var  = stats[C + c] * inv - mean * mean;
  float sc   = g[c] * rsqrtf(var + 1e-5f);
  float sh   = b[c] - mean * sc;
  int r0 = blockIdx.x * BN_ROWS;
  int r1 = r0 + BN_ROWS; if (r1 > M) r1 = M;
  for (int rr = r0; rr < r1; ++rr) {
    float v = fmaf(y[(size_t)rr * C + c], sc, sh);
    o[(size_t)rr * C + c] = fmaxf(v, 0.f);
  }
}

// ---------------- host-side orchestration ----------------

static inline void launch_gemm(const _Float16* A, const _Float16* B, float* C,
                               int M, int N, int K, hipStream_t s) {
  int numMT  = (M + 15) >> 4;
  int numNT  = N >> 6;
  int waves  = numMT * numNT;
  int blocks = (waves + 7) / 8;  // 8 waves / 256-thread block
  wmma_gemm_k<<<blocks, 256, 0, s>>>(A, B, C, M, N, K, numNT);
}

static inline void launch_zero(float* p, long long n, hipStream_t s) {
  long long b = (n + 255) / 256;
  int blocks = (int)(b > 4096 ? 4096 : (b < 1 ? 1 : b));
  zero_f32_k<<<blocks, 256, 0, s>>>(p, n);
}

static inline void launch_cvt(const float* src, _Float16* dst, int n, hipStream_t s) {
  cvt_f16_k<<<(n + 255) / 256, 256, 0, s>>>(src, dst, n);
}

extern "C" void kernel_launch(void* const* d_in, const int* in_sizes, int n_in,
                              void* d_out, int out_size, void* d_ws, size_t ws_size,
                              hipStream_t stream) {
  (void)in_sizes; (void)n_in; (void)out_size; (void)ws_size;
  const int RE = 60000;

  const float* edge_rep = (const float*)d_in[0];
  const float* rep[6] = {(const float*)d_in[1], (const float*)d_in[2], (const float*)d_in[3],
                         (const float*)d_in[4], (const float*)d_in[5], (const float*)d_in[6]};
  const float* Wab[6];
  for (int i = 0; i < 6; ++i) Wab[i] = (const float*)d_in[7 + i];

  const float* eW1 = (const float*)d_in[13];
  const float* eg1 = (const float*)d_in[14];
  const float* eb1 = (const float*)d_in[15];
  const float* eW2 = (const float*)d_in[16];
  const float* eg2 = (const float*)d_in[17];
  const float* eb2 = (const float*)d_in[18];
  const float* cW1 = (const float*)d_in[19];
  const float* cg1 = (const float*)d_in[20];
  const float* cb1 = (const float*)d_in[21];
  const float* cW2 = (const float*)d_in[22];
  const float* cg2 = (const float*)d_in[23];
  const float* cb2 = (const float*)d_in[24];
  const float* pW1 = (const float*)d_in[25];
  const float* pg1 = (const float*)d_in[26];
  const float* pb1 = (const float*)d_in[27];
  const float* pW2 = (const float*)d_in[28];
  const float* pg2 = (const float*)d_in[29];
  const float* pb2 = (const float*)d_in[30];

  // setup_inputs dict order interleaves e2x / x2e per family
  const int* e2s[6] = {(const int*)d_in[31], (const int*)d_in[33], (const int*)d_in[35],
                       (const int*)d_in[37], (const int*)d_in[39], (const int*)d_in[41]};
  const int* s2e[6] = {(const int*)d_in[32], (const int*)d_in[34], (const int*)d_in[36],
                       (const int*)d_in[38], (const int*)d_in[40], (const int*)d_in[42]};
  const int rF[6]    = {60000, 72000, 60000, 75000, 90000, 105000};
  const int kF[6]    = {5, 6, 4, 5, 6, 7};
  const int mE2S[6]  = {120000, 144000, 120000, 150000, 180000, 210000};
  const long long outoff[6] = {60000, 120000, 192000, 252000, 327000, 417000};

  // ---- workspace carve (256B aligned by construction) ----
  char* ws = (char*)d_ws;
  float* A_c = (float*)(ws);                       // 60000*256*4 = 61,440,000
  float* A_p = (float*)(ws + 61440000);            // 61,440,000
  _Float16* w16 = (_Float16*)(ws + 122880000);     // 1,441,792 bytes of f16 weights
  _Float16* wab16[6];
  for (int i = 0; i < 6; ++i) wab16[i] = w16 + (size_t)i * 32768;
  _Float16* cW1_16 = w16 + 196608;
  _Float16* cW2_16 = cW1_16 + 65536;
  _Float16* pW1_16 = cW2_16 + 32768;
  _Float16* pW2_16 = pW1_16 + 65536;
  _Float16* eW1_16 = pW2_16 + 32768;
  _Float16* eW2_16 = eW1_16 + 294912;
  float* stats = (float*)(ws + 124321792);         // 512 floats
  char* scr = ws + 124323840;                      // 276,480,000-byte scratch arena
  float*    buf_a  = (float*)(scr);                //  53,760,000  (105000x128 f32)
  float*    buf_gs = (float*)(scr + 53760000);     //   7,680,000  (15000x128 f32)
  _Float16* g16    = (_Float16*)(scr + 61440000);  //  53,760,000  (105000x256 f16)
  float*    gw     = (float*)(scr + 115200000);    //  53,760,000  (105000x128 f32; reused as y2)
  _Float16* n16    = (_Float16*)(scr + 168960000); //  53,760,000  (105000x256 f16)
  _Float16* h16    = (_Float16*)(scr + 222720000); //  53,760,000  (105000x256 f16)
  float*    ybuf   = (float*)(scr);                // alias over buf_a+gs+g16 (105000x256 f32)
  // edge phase aliases (family buffers dead by then)
  _Float16* XE   = (_Float16*)(scr);               // 60000x1152 f16 = 138,240,000
  float*    yE   = (float*)(scr + 168960000);      // 60000x256 f32  =  61,440,000
  _Float16* h16E = (_Float16*)(scr + 230400000);   // 60000x256 f16  =  30,720,000
  float*    y2E  = (float*)(scr);                  // 60000x128 f32 (XE dead)

  // ---- convert weights to f16 (B operands) ----
  for (int i = 0; i < 6; ++i) launch_cvt(Wab[i], wab16[i], 32768, stream);
  launch_cvt(cW1, cW1_16, 65536, stream);
  launch_cvt(cW2, cW2_16, 32768, stream);
  launch_cvt(pW1, pW1_16, 65536, stream);
  launch_cvt(pW2, pW2_16, 32768, stream);
  launch_cvt(eW1, eW1_16, 294912, stream);
  launch_cvt(eW2, eW2_16, 32768, stream);

  launch_zero(A_c, (long long)RE * 256, stream);
  launch_zero(A_p, (long long)RE * 256, stream);

  // ---- six subgraph families ----
  for (int f = 0; f < 6; ++f) {
    int R = rF[f], K = kF[f], G = R / K;
    float* Afam = (f < 2) ? A_c : A_p;
    const _Float16* W1_16 = (f < 2) ? cW1_16 : pW1_16;
    const _Float16* W2_16 = (f < 2) ? cW2_16 : pW2_16;
    const float* g1 = (f < 2) ? cg1 : pg1;
    const float* b1 = (f < 2) ? cb1 : pb1;
    const float* g2 = (f < 2) ? cg2 : pg2;
    const float* b2 = (f < 2) ? cb2 : pb2;

    // gather1(edge_rep) = [a | group-sum broadcast]
    launch_zero(buf_a, (long long)R * 128, stream);
    scatter_add128_k<<<mE2S[f], 128, 0, stream>>>(edge_rep, e2s[f], mE2S[f], buf_a);
    group_sum_k<<<G, 128, 0, stream>>>(buf_a, buf_gs, K);
    pack_g16_k<<<R, 256, 0, stream>>>(buf_a, buf_gs, g16, K);

    // Autobahn GEMM: g[R,256] @ W_ab[256,128] -> gw[R,128]
    launch_gemm(g16, wab16[f], gw, R, 128, 256, stream);

    // new = [gw | rep]; scatter back to per-family edge accumulator
    pack_new16_k<<<R, 256, 0, stream>>>(gw, rep[f], n16);
    scatter_add_new_k<<<2 * RE, 256, 0, stream>>>(gw, rep[f], s2e[f], 2 * RE, Afam);

    // MLP layer 1: new[R,256] @ W1[256,256] -> BN -> relu -> h16
    launch_zero(stats, 512, stream);
    launch_gemm(n16, W1_16, ybuf, R, 256, 256, stream);
    int nb = (R + BN_ROWS - 1) / BN_ROWS;
    bn_stats_k<<<nb, 256, 0, stream>>>(ybuf, stats, R, 256);
    bn_relu_f16_k<<<nb, 256, 0, stream>>>(ybuf, stats, g1, b1, h16, R, 256);

    // MLP layer 2: h16[R,256] @ W2[256,128] -> BN -> relu -> d_out rows
    launch_zero(stats, 512, stream);
    launch_gemm(h16, W2_16, gw, R, 128, 256, stream);
    bn_stats_k<<<nb, 128, 0, stream>>>(gw, stats, R, 128);
    bn_relu_f32_k<<<nb, 128, 0, stream>>>(gw, stats, g2, b2,
                                          (float*)d_out + outoff[f] * 128, R, 128);
  }

  // ---- edge MLP: X = [edge | A_c | pairsum(A_c) | A_p | pairsum(A_p)] (9H = 1152) ----
  build_xe_k<<<RE, 128, 0, stream>>>(edge_rep, A_c, A_p, XE);

  launch_zero(stats, 512, stream);
  launch_gemm(XE, eW1_16, yE, RE, 256, 1152, stream);
  int nbE = (RE + BN_ROWS - 1) / BN_ROWS;
  bn_stats_k<<<nbE, 256, 0, stream>>>(yE, stats, RE, 256);
  bn_relu_f16_k<<<nbE, 256, 0, stream>>>(yE, stats, eg1, eb1, h16E, RE, 256);

  launch_zero(stats, 512, stream);
  launch_gemm(h16E, eW2_16, y2E, RE, 128, 256, stream);
  bn_stats_k<<<nbE, 128, 0, stream>>>(y2E, stats, RE, 128);
  bn_relu_f32_k<<<nbE, 128, 0, stream>>>(y2E, stats, eg2, eb2, (float*)d_out, RE, 128);
}
